// GPT2_17884243821073
// MI455X (gfx1250) — matrix operand order repaired and checked by reading, not compiled, not run on hardware
//
#include <hip/hip_runtime.h>
#include <hip/hip_bf16.h>

// ---------------- CDNA5 WMMA types ----------------
typedef __attribute__((ext_vector_type(16))) __bf16 v16bf;
typedef __attribute__((ext_vector_type(8)))  float  v8f;

static constexpr int Ecnt = 1024, Hn = 16, Ln = 12, Vn = 32000, FFn = 4096;
static constexpr int Bn = 2, Sn = 2048, Dn = 64, Rows = Bn * Sn; // 4096

__device__ __forceinline__ v8f wmma_bf16(v16bf a, v16bf b, v8f c) {
    // (neg_a, A, neg_b, B, c_mod, C, reuse_a, reuse_b)
    return __builtin_amdgcn_wmma_f32_16x16x32_bf16(false, a, false, b, (short)0, c,
                                                   false, false);
}

union FragU { uint4 u[2]; v16bf v; };

// max across each 16-lane row group (lanes 0-15 / 16-31 independently).
#if __has_builtin(__builtin_amdgcn_update_dpp)
template <int CTRL>
__device__ __forceinline__ float dpp_mov(float x) {
    return __builtin_bit_cast(float, __builtin_amdgcn_update_dpp(
        0, __builtin_bit_cast(int, x), CTRL, 0xF, 0xF, true));
}
__device__ __forceinline__ float rowmax16(float x) {
    x = fmaxf(x, dpp_mov<0xB1>(x));   // quad_perm [1,0,3,2]  (xor 1)
    x = fmaxf(x, dpp_mov<0x4E>(x));   // quad_perm [2,3,0,1]  (xor 2)
    x = fmaxf(x, dpp_mov<0x124>(x));  // row_ror:4
    x = fmaxf(x, dpp_mov<0x128>(x));  // row_ror:8
    return x;
}
#else
__device__ __forceinline__ float rowmax16(float x) {
    for (int off = 1; off < 16; off <<= 1) x = fmaxf(x, __shfl_xor(x, off, 32));
    return x;
}
#endif

// A fragment: 16x32 bf16, lane m = lane%16, khalf = lane/16.
// lanes 0-15: K = {0..7, 16..23}; lanes 16-31: K = {8..15, 24..31}
__device__ __forceinline__ v16bf load_afrag(const __bf16* A, int lda, int row,
                                            int k0, int khalf) {
    const __bf16* p = A + (size_t)row * lda + k0;
    FragU t;
    t.u[0] = *(const uint4*)(p + khalf * 8);
    t.u[1] = *(const uint4*)(p + 16 + khalf * 8);
    return t.v;
}

// B fragment from BT[N][K] (row = output col n): lanes 0-15 K={0..7,16..23}, lanes 16-31 K={8..15,24..31}
__device__ __forceinline__ v16bf load_bfrag(const __bf16* BT, int ldb, int nrow,
                                            int k0, int khalf) {
    const __bf16* p = BT + (size_t)nrow * ldb + k0 + khalf * 8;
    FragU t;
    t.u[0] = *(const uint4*)(p);
    t.u[1] = *(const uint4*)(p + 16);
    return t.v;
}

// CDNA5 async global->LDS copy of 16 bytes (per-lane addresses), tracked by ASYNCcnt.
__device__ __forceinline__ void async_copy_b128(unsigned ldsOff, const void* gptr) {
    asm volatile("global_load_async_to_lds_b128 %0, %1, off"
                 :: "v"(ldsOff), "v"((unsigned long long)(size_t)gptr)
                 : "memory");
}

// ---------------- embedding: h = tok_emb[x] + pe[b] (quirk: pe indexed by batch) ----
__global__ void embed_kernel(const int* __restrict__ x, const float* __restrict__ tok,
                             const float* __restrict__ pe, float* __restrict__ h) {
    int row = blockIdx.x;            // 0..4095
    int b = row >> 11;               // row / S
    int t = x[row];
    int e = threadIdx.x * 4;
    float4 a = *(const float4*)(tok + (size_t)t * Ecnt + e);
    float4 p = *(const float4*)(pe + (size_t)b * Ecnt + e);
    float4 o = {a.x + p.x, a.y + p.y, a.z + p.z, a.w + p.w};
    *(float4*)(h + (size_t)row * Ecnt + e) = o;
}

// ---------------- layernorm over E=1024; 256 threads/row --------------------------
// MODE 0: write bf16 + transposed bf16 [b][h][d][s]   (pre-attention)
// MODE 1: write fp32 + bf16                            (pre-FFN)
template <int MODE>
__global__ void ln_kernel(const float* __restrict__ in, const float* __restrict__ g,
                          const float* __restrict__ be, float* __restrict__ outF,
                          __bf16* __restrict__ outB, __bf16* __restrict__ outT) {
    __shared__ float red[16];
    int row = blockIdx.x;
    int tid = threadIdx.x, lane = tid & 31, w = tid >> 5;
    float4 v = *(const float4*)(in + (size_t)row * Ecnt + tid * 4);
    float s1 = v.x + v.y + v.z + v.w;
    float s2 = v.x * v.x + v.y * v.y + v.z * v.z + v.w * v.w;
    for (int off = 16; off >= 1; off >>= 1) {
        s1 += __shfl_xor(s1, off, 32);
        s2 += __shfl_xor(s2, off, 32);
    }
    if (lane == 0) { red[w] = s1; red[8 + w] = s2; }
    __syncthreads();
    if (tid == 0) {
        float a = 0.f, bb = 0.f;
        for (int i = 0; i < 8; ++i) { a += red[i]; bb += red[8 + i]; }
        red[0] = a; red[8] = bb;
    }
    __syncthreads();
    float mean = red[0] * (1.f / Ecnt);
    float var  = red[8] * (1.f / Ecnt) - mean * mean;
    float rstd = rsqrtf(var + 1e-5f);
    float vv[4] = {v.x, v.y, v.z, v.w};
#pragma unroll
    for (int j = 0; j < 4; ++j) {
        int e = tid * 4 + j;
        float y = (vv[j] - mean) * rstd * g[e] + be[e];
        if (MODE == 1) {
            outF[(size_t)row * Ecnt + e] = y;
            outB[(size_t)row * Ecnt + e] = (__bf16)y;
        } else {
            outB[(size_t)row * Ecnt + e] = (__bf16)y;
            int hh = e >> 6, d = e & 63, bb = row >> 11, ss = row & (Sn - 1);
            outT[((size_t)((bb * Hn + hh) * Dn + d)) * Sn + ss] = (__bf16)y;
        }
    }
}

// ---------------- fp32 [R,C] -> bf16 [C,R] transpose (weight prep) ----------------
__global__ void transpose_cvt(const float* __restrict__ in, __bf16* __restrict__ out,
                              int R, int C) {
    __shared__ float tile[32][33];
    int c0 = blockIdx.x * 32, r0 = blockIdx.y * 32;
#pragma unroll
    for (int i = 0; i < 4; ++i) {
        int r = r0 + threadIdx.y + i * 8;
        tile[threadIdx.y + i * 8][threadIdx.x] = in[(size_t)r * C + c0 + threadIdx.x];
    }
    __syncthreads();
#pragma unroll
    for (int i = 0; i < 4; ++i) {
        int c = c0 + threadIdx.y + i * 8;
        out[(size_t)c * R + r0 + threadIdx.x] =
            (__bf16)tile[threadIdx.x][threadIdx.y + i * 8];
    }
}

// ---------------- fp32 -> bf16 elementwise ---------------------------------------
__global__ void cvt_bf16(const float* __restrict__ in, __bf16* __restrict__ out) {
    size_t i = ((size_t)blockIdx.x * blockDim.x + threadIdx.x) * 4;
#pragma unroll
    for (int j = 0; j < 4; ++j) out[i + j] = (__bf16)in[i + j];
}

// ---------------- bf16 GEMM with async double-buffered LDS staging ----------------
// C = A[M,K] * BT[N,K]^T + bias (+resid)(+relu)
// block = 256 thr = 8 waves (4x2), wave tile = 32x64, block tile = 128x128.
// Per 32-wide K step both 128x32 tiles are streamed into LDS with
// global_load_async_to_lds_b128 (ASYNCcnt). Single barrier per step: at the barrier
// every wave has finished reading the other stage (program order) and waited for its
// own stage-s copies, so issuing the next stage after the barrier cannot race.
template <int RELU>
__global__ void gemm_bf16(const __bf16* __restrict__ A, const __bf16* __restrict__ BT,
                          const float* __restrict__ bias, const float* __restrict__ resid,
                          float* __restrict__ outF, __bf16* __restrict__ outB,
                          int M, int N, int K) {
    __shared__ __align__(16) __bf16 Ash[2][128][40];
    __shared__ __align__(16) __bf16 Bsh[2][128][40];
    int tid = threadIdx.x;
    int lane = tid & 31, w = tid >> 5;
    int ln = lane & 15, kh = lane >> 4;
    int wm = w >> 1, wn = w & 1;
    int mblk = blockIdx.y * 128, nblk = blockIdx.x * 128;

    // staging: 512 chunks of 16B per tile; thread t owns chunks t and t+256
    int r0c = tid >> 2, p0c = tid & 3;           // chunk tid
    int r1c = (tid + 256) >> 2, p1c = tid & 3;   // chunk tid+256
    auto stage = [&](int s, int k0) {
        async_copy_b128((unsigned)(size_t)&Ash[s][r0c][p0c * 8],
                        A + (size_t)(mblk + r0c) * K + k0 + p0c * 8);
        async_copy_b128((unsigned)(size_t)&Ash[s][r1c][p1c * 8],
                        A + (size_t)(mblk + r1c) * K + k0 + p1c * 8);
        async_copy_b128((unsigned)(size_t)&Bsh[s][r0c][p0c * 8],
                        BT + (size_t)(nblk + r0c) * K + k0 + p0c * 8);
        async_copy_b128((unsigned)(size_t)&Bsh[s][r1c][p1c * 8],
                        BT + (size_t)(nblk + r1c) * K + k0 + p1c * 8);
    };

    v8f acc[2][4];
#pragma unroll
    for (int mi = 0; mi < 2; ++mi)
#pragma unroll
        for (int f = 0; f < 4; ++f)
#pragma unroll
            for (int r = 0; r < 8; ++r) acc[mi][f][r] = 0.f;

    int nsteps = K >> 5;
    stage(0, 0);
    for (int i = 0; i < nsteps; ++i) {
        int s = i & 1;
        asm volatile("s_wait_asynccnt 0x0" ::: "memory"); // my stage-s copies arrived
        __syncthreads();   // everyone's copies visible; everyone done reading s^1
        if (i + 1 < nsteps) stage(s ^ 1, (i + 1) * 32);   // next tile in flight

        FragU a0, a1, b[4];
        a0.u[0] = *(const uint4*)&Ash[s][wm * 32 + ln][kh * 8];
        a0.u[1] = *(const uint4*)&Ash[s][wm * 32 + ln][16 + kh * 8];
        a1.u[0] = *(const uint4*)&Ash[s][wm * 32 + 16 + ln][kh * 8];
        a1.u[1] = *(const uint4*)&Ash[s][wm * 32 + 16 + ln][16 + kh * 8];
#pragma unroll
        for (int f = 0; f < 4; ++f) {
            b[f].u[0] = *(const uint4*)&Bsh[s][wn * 64 + f * 16 + ln][kh * 8];
            b[f].u[1] = *(const uint4*)&Bsh[s][wn * 64 + f * 16 + ln][16 + kh * 8];
        }
#pragma unroll
        for (int f = 0; f < 4; ++f) {
            acc[0][f] = wmma_bf16(a0.v, b[f].v, acc[0][f]);
            acc[1][f] = wmma_bf16(a1.v, b[f].v, acc[1][f]);
        }
    }
#pragma unroll
    for (int mi = 0; mi < 2; ++mi)
#pragma unroll
        for (int f = 0; f < 4; ++f)
#pragma unroll
            for (int r = 0; r < 8; ++r) {
                int row = mblk + wm * 32 + mi * 16 + 8 * kh + r;
                int col = nblk + wn * 64 + f * 16 + ln;
                float v = acc[mi][f][r] + bias[col];
                if (resid) v += resid[(size_t)row * N + col];
                if (RELU) v = fmaxf(v, 0.f);
                if (outF) outF[(size_t)row * N + col] = v;
                if (outB) outB[(size_t)row * N + col] = (__bf16)v;
            }
}

// ---------------- flash attention: q=k=v=hn, causal, scale = 1/sqrt(E) ------------
// grid (B*H, S/16/8), 8 waves/block, 1 wave = 16 q rows.
// Row max via DPP (VALU only); row sums of exp(P) via one WMMA against an all-ones
// B fragment (C layout == per-lane row mapping, zero shuffles). V^T fragments are
// loaded right after the P LDS stores: short live range (no spills), latency covered
// by the pa reads + sum-WMMA + bookkeeping; prefetches warm the next tile.
__global__ void __launch_bounds__(256, 1)
attn_kernel(const __bf16* __restrict__ hn, const __bf16* __restrict__ hnT,
            __bf16* __restrict__ attn_out) {
    __shared__ __align__(16) __bf16 Pl[8][16][32];
    int lane = threadIdx.x & 31, w = threadIdx.x >> 5;
    int ln = lane & 15, kh = lane >> 4;
    int bh = blockIdx.x, b = bh >> 4, hd = bh & 15;
    int q0 = (blockIdx.y * 8 + w) * 16;
    const float scale = 1.f / 32.f; // 1/sqrt(1024)

    v16bf qf[2];
#pragma unroll
    for (int c = 0; c < 2; ++c)
        qf[c] = load_afrag(hn, Ecnt, b * Sn + q0 + ln, hd * Dn + c * 32, kh);

    FragU ones;   // all-ones 32x16 bf16 B fragment (1.0bf16 = 0x3F80)
    ones.u[0] = ones.u[1] = uint4{0x3F803F80u, 0x3F803F80u, 0x3F803F80u, 0x3F803F80u};

    v8f o[4];
    float mrun[8], lrun[8];
#pragma unroll
    for (int f = 0; f < 4; ++f)
#pragma unroll
        for (int r = 0; r < 8; ++r) o[f][r] = 0.f;
#pragma unroll
    for (int r = 0; r < 8; ++r) { mrun[r] = -1e30f; lrun[r] = 0.f; }

    for (int k0 = 0; k0 < q0 + 16; k0 += 32) {
        float p[2][8], tmax[8];
#pragma unroll
        for (int r = 0; r < 8; ++r) tmax[r] = -1e30f;
#pragma unroll
        for (int t = 0; t < 2; ++t) {
            v8f s;
#pragma unroll
            for (int r = 0; r < 8; ++r) s[r] = 0.f;
#pragma unroll
            for (int c = 0; c < 2; ++c) {
                v16bf kb = load_bfrag(hn, Ecnt, b * Sn + k0 + t * 16 + ln,
                                      hd * Dn + c * 32, kh);
                s = wmma_bf16(qf[c], kb, s);
            }
            int col = k0 + t * 16 + ln;
#pragma unroll
            for (int r = 0; r < 8; ++r) {
                int row = q0 + 8 * kh + r;
                float v = (col <= row) ? s[r] * scale : -1e30f;
                p[t][r] = v;
                tmax[r] = fmaxf(tmax[r], v);
            }
        }
        // warm next tile / this tile's V^T rows while softmax runs (no reg cost)
        __builtin_prefetch(hnT + (size_t)(bh * Dn + ln) * Sn + k0, 0, 3);
        if (k0 + 32 < q0 + 16)
            __builtin_prefetch(hn + (size_t)(b * Sn + k0 + 32 + ln) * Ecnt + hd * Dn, 0, 3);

#pragma unroll
        for (int r = 0; r < 8; ++r) tmax[r] = rowmax16(tmax[r]);

        float corr[8];
#pragma unroll
        for (int r = 0; r < 8; ++r) {
            float mn = fmaxf(mrun[r], tmax[r]);
            corr[r] = __expf(mrun[r] - mn);
            mrun[r] = mn;
        }
#pragma unroll
        for (int t = 0; t < 2; ++t)
#pragma unroll
            for (int r = 0; r < 8; ++r) p[t][r] = __expf(p[t][r] - mrun[r]);
#pragma unroll
        for (int r = 0; r < 8; ++r) lrun[r] *= corr[r];
#pragma unroll
        for (int f = 0; f < 4; ++f)
#pragma unroll
            for (int r = 0; r < 8; ++r) o[f][r] *= corr[r];

        // P (C-layout) -> LDS -> A-fragment layout. Wave-private patch; LDS ops from
        // one wave execute in order, just stop compiler reordering.
#pragma unroll
        for (int t = 0; t < 2; ++t)
#pragma unroll
            for (int r = 0; r < 8; ++r) Pl[w][8 * kh + r][t * 16 + ln] = (__bf16)p[t][r];

        // V^T fragments: short live range, overlapped with pa reads + sum-WMMA
        v16bf vb[4];
#pragma unroll
        for (int f = 0; f < 4; ++f)
            vb[f] = load_bfrag(hnT, Sn, bh * Dn + f * 16 + ln, k0, kh);

        asm volatile("" ::: "memory");
        FragU pa;
        pa.u[0] = *(const uint4*)&Pl[w][ln][kh * 8];
        pa.u[1] = *(const uint4*)&Pl[w][ln][16 + kh * 8];

        v8f ltf;
#pragma unroll
        for (int r = 0; r < 8; ++r) ltf[r] = 0.f;
        ltf = wmma_bf16(pa.v, ones.v, ltf);          // row sums of exp(P)
#pragma unroll
        for (int f = 0; f < 4; ++f) o[f] = wmma_bf16(pa.v, vb[f], o[f]);
#pragma unroll
        for (int r = 0; r < 8; ++r) lrun[r] += ltf[r];
    }
#pragma unroll
    for (int r = 0; r < 8; ++r) lrun[r] = 1.f / lrun[r];
#pragma unroll
    for (int f = 0; f < 4; ++f)
#pragma unroll
        for (int r = 0; r < 8; ++r) {
            size_t row = (size_t)(b * Sn + q0 + 8 * kh + r);
            attn_out[row * Ecnt + hd * Dn + f * 16 + ln] = (__bf16)(o[f][r] * lrun[r]);
        }
}

// ---------------- driver ----------------------------------------------------------
extern "C" void kernel_launch(void* const* d_in, const int* in_sizes, int n_in,
                              void* d_out, int out_size, void* d_ws, size_t ws_size,
                              hipStream_t stream) {
    const int*   x       = (const int*)d_in[0];
    const float* tok_emb = (const float*)d_in[2];
    const float* pe      = (const float*)d_in[3];
    const float* ln1_g   = (const float*)d_in[4];
    const float* ln1_b   = (const float*)d_in[5];
    const float* ln2_g   = (const float*)d_in[6];
    const float* ln2_b   = (const float*)d_in[7];
    const float* wo      = (const float*)d_in[8];
    const float* bo      = (const float*)d_in[9];
    const float* w1      = (const float*)d_in[10];
    const float* b1      = (const float*)d_in[11];
    const float* w2      = (const float*)d_in[12];
    const float* b2      = (const float*)d_in[13];
    const float* out_w   = (const float*)d_in[14];
    const float* out_b   = (const float*)d_in[15];
    float* out = (float*)d_out;

    char* ws = (char*)d_ws;
    size_t off = 0;
    auto alloc = [&](size_t bytes) {
        void* p = ws + off;
        off += (bytes + 255) & ~(size_t)255;
        return p;
    };
    float*  h_f   = (float*)alloc((size_t)Rows * Ecnt * 4);
    float*  x2_f  = (float*)alloc((size_t)Rows * Ecnt * 4);
    float*  h2_f  = (float*)alloc((size_t)Rows * Ecnt * 4);
    __bf16* hn_b  = (__bf16*)alloc((size_t)Rows * Ecnt * 2);
    __bf16* hnT_b = (__bf16*)alloc((size_t)Rows * Ecnt * 2);
    __bf16* h2_b  = (__bf16*)alloc((size_t)Rows * Ecnt * 2);
    __bf16* at_b  = (__bf16*)alloc((size_t)Rows * Ecnt * 2);
    __bf16* r1_b  = (__bf16*)alloc((size_t)Rows * FFn * 2);
    __bf16* hf_b  = (__bf16*)alloc((size_t)Rows * Ecnt * 2);
    __bf16* woT   = (__bf16*)alloc((size_t)Ecnt * Ecnt * 2);
    __bf16* w1T   = (__bf16*)alloc((size_t)Ecnt * FFn * 2);
    __bf16* w2T   = (__bf16*)alloc((size_t)FFn * Ecnt * 2);
    __bf16* owT   = (__bf16*)alloc((size_t)Ecnt * Vn * 2);

    dim3 t32x8(32, 8);

    // embedding + vocab-weight prep
    embed_kernel<<<Rows, 256, 0, stream>>>(x, tok_emb, pe, h_f);
    transpose_cvt<<<dim3(Vn / 32, Ecnt / 32), t32x8, 0, stream>>>(out_w, owT, Ecnt, Vn);

    for (int l = 0; l < Ln; ++l) {
        ln_kernel<0><<<Rows, 256, 0, stream>>>(h_f, ln1_g + l * Ecnt, ln1_b + l * Ecnt,
                                               nullptr, hn_b, hnT_b);
        attn_kernel<<<dim3(Bn * Hn, Sn / 128), 256, 0, stream>>>(hn_b, hnT_b, at_b);

        transpose_cvt<<<dim3(Ecnt / 32, Ecnt / 32), t32x8, 0, stream>>>(
            wo + (size_t)l * Ecnt * Ecnt, woT, Ecnt, Ecnt);
        gemm_bf16<0><<<dim3(Ecnt / 128, Rows / 128), 256, 0, stream>>>(
            at_b, woT, bo + l * Ecnt, h_f, x2_f, nullptr, Rows, Ecnt, Ecnt);

        ln_kernel<1><<<Rows, 256, 0, stream>>>(x2_f, ln2_g + l * Ecnt, ln2_b + l * Ecnt,
                                               h2_f, h2_b, nullptr);

        transpose_cvt<<<dim3(FFn / 32, Ecnt / 32), t32x8, 0, stream>>>(
            w1 + (size_t)l * Ecnt * FFn, w1T, Ecnt, FFn);
        gemm_bf16<1><<<dim3(FFn / 128, Rows / 128), 256, 0, stream>>>(
            h2_b, w1T, b1 + l * FFn, nullptr, nullptr, r1_b, Rows, FFn, Ecnt);

        transpose_cvt<<<dim3(Ecnt / 32, FFn / 32), t32x8, 0, stream>>>(
            w2 + (size_t)l * FFn * Ecnt, w2T, FFn, Ecnt);
        gemm_bf16<0><<<dim3(Ecnt / 128, Rows / 128), 256, 0, stream>>>(
            r1_b, w2T, b2 + l * Ecnt, h2_f, h_f, nullptr, Rows, Ecnt, FFn);
    }

    cvt_bf16<<<Rows * Ecnt / 1024, 256, 0, stream>>>(h_f, hf_b);
    gemm_bf16<0><<<dim3(Vn / 128, Rows / 128), 256, 0, stream>>>(
        hf_b, owT, out_b, nullptr, out, nullptr, Rows, Vn, Ecnt);
}
